// PTBlock_2396591751782
// MI455X (gfx1250) — matrix operand order repaired and checked by reading, not compiled, not run on hardware
//
#include <hip/hip_runtime.h>
#include <hip/hip_bf16.h>

#define NPTS 8192
#define CH   64
#define KNN  16
#define BATCH 2

typedef __attribute__((ext_vector_type(16))) _Float16 v16h;
typedef __attribute__((ext_vector_type(8)))  float    v8f;

#define LDS_FENCE() asm volatile("s_wait_dscnt 0" ::: "memory")

__device__ __forceinline__ v8f vzero8() {
  v8f z = {0.f,0.f,0.f,0.f,0.f,0.f,0.f,0.f};
  return z;
}

// A-fragment: 16x32 f16 from row-major [64][64] LDS weight matrix.
// Lane L (0-15): row M=L, K = {kb+0..7, 16+kb+0..7} with kb=0; lanes 16-31: kb=8.
__device__ __forceinline__ v16h load_a_frag(const _Float16* W, int rt, int kc, int lane) {
  int row = rt * 16 + (lane & 15);
  int kb  = (lane & 16) ? 8 : 0;
  const _Float16* p0 = W + row * 64 + kc * 32 + kb;
  const _Float16* p1 = p0 + 16;
  v16h a;
#pragma unroll
  for (int h = 0; h < 8; ++h) { a[h] = p0[h]; a[h + 8] = p1[h]; }
  return a;
}

// B-fragment: 32x16 f16 from column-major [cols][64] LDS buffer.
// Lane L: column N = L&15; K = {kb+0..7, 16+kb+0..7} with kb = (L<16) ? 0 : 8 within the k-chunk.
__device__ __forceinline__ v16h load_b_frag(const _Float16* buf, int col, int kc, int lane) {
  const _Float16* p = buf + col * 64 + kc * 32 + ((lane & 16) ? 8 : 0);
  v16h b;
#pragma unroll
  for (int h = 0; h < 16; ++h) b[h] = p[(h < 8) ? h : h + 8];
  return b;
}

// D = W[64x64] * X[64x16]: 4 row-tiles x 2 k-chunks = 8 WMMAs.
__device__ __forceinline__ void gemm64x16(const _Float16* Wsm, const _Float16* colbuf,
                                          int colbase, int lane, v8f acc[4]) {
  int col = colbase + (lane & 15);
  v16h b0 = load_b_frag(colbuf, col, 0, lane);
  v16h b1 = load_b_frag(colbuf, col, 1, lane);
#pragma unroll
  for (int rt = 0; rt < 4; ++rt) {
    v16h a0 = load_a_frag(Wsm, rt, 0, lane);
    v16h a1 = load_a_frag(Wsm, rt, 1, lane);
    acc[rt] = __builtin_amdgcn_wmma_f32_16x16x32_f16(false, a0, false, b0, (short)0, acc[rt], false, false);
    acc[rt] = __builtin_amdgcn_wmma_f32_16x16x32_f16(false, a1, false, b1, (short)0, acc[rt], false, false);
  }
}

// ---------------------------------------------------------------------------
// Kernel 1: kNN (top-16 nearest incl. self). One wave per row, 32 rows/wave.
// Points staged fully in LDS (96 KB of 320 KB WGP LDS).
// ---------------------------------------------------------------------------
__global__ void knn_kernel(const float* __restrict__ p, int* __restrict__ idxw) {
  extern __shared__ char smem[];
  float* pl = (float*)smem;  // [3*NPTS]
  int tid = threadIdx.x, lane = tid & 31, wv = tid >> 5;
  int b = blockIdx.x >> 5, blk = blockIdx.x & 31;
  const float* pb = p + (size_t)b * 3 * NPTS;
  for (int i = tid; i < 3 * NPTS; i += 256) pl[i] = pb[i];
  __syncthreads();
  const float* px = pl;
  const float* py = pl + NPTS;
  const float* pz = pl + 2 * NPTS;

  for (int r = 0; r < 32; ++r) {
    int n = blk * 256 + wv * 32 + r;
    float xn = px[n], yn = py[n], zn = pz[n];
    float vals[16]; int ids[16];
#pragma unroll
    for (int j = 0; j < 16; ++j) { vals[j] = 3.4e38f; ids[j] = 0; }
    float worst = 3.4e38f; int wslot = 0;
    for (int m = lane; m < NPTS; m += 32) {
      float dx = px[m] - xn, dy = py[m] - yn, dz = pz[m] - zn;
      float d = dx * dx + dy * dy + dz * dz;
      if (d < worst) {
        vals[wslot] = d; ids[wslot] = m;
        worst = vals[0]; wslot = 0;
#pragma unroll
        for (int j = 1; j < 16; ++j) if (vals[j] > worst) { worst = vals[j]; wslot = j; }
      }
    }
    int* orow = idxw + ((size_t)b * NPTS + n) * KNN;
    for (int k = 0; k < KNN; ++k) {
      float bv = vals[0]; int bi = ids[0]; int bs = 0;
#pragma unroll
      for (int j = 1; j < 16; ++j)
        if (vals[j] < bv || (vals[j] == bv && ids[j] < bi)) { bv = vals[j]; bi = ids[j]; bs = j; }
      float mv = bv; int mi = bi;
#pragma unroll
      for (int s = 16; s >= 1; s >>= 1) {
        float ov = __shfl_xor(mv, s, 32);
        int   oi = __shfl_xor(mi, s, 32);
        if (ov < mv || (ov == mv && oi < mi)) { mv = ov; mi = oi; }
      }
      if (mi == bi) vals[bs] = 3.4e38f;   // lane subsets disjoint -> unique owner
      if (lane == k) orow[k] = mi;
    }
  }
}

// ---------------------------------------------------------------------------
// Kernel 2: x = bn(W_top@x_in); phi/psi/alpha = W@x + b, stored TRANSPOSED
// [B,N,64] so later neighbor gathers are 256B-contiguous. 1 tile (16 cols)
// per wave; 32 WMMAs per wave.
// ---------------------------------------------------------------------------
__global__ void feat_kernel(const float* __restrict__ x_in,
                            const float* __restrict__ W_top, const float* __restrict__ b_top,
                            const float* __restrict__ s_top, const float* __restrict__ t_top,
                            const float* __restrict__ W_phi, const float* __restrict__ b_phi,
                            const float* __restrict__ W_psi, const float* __restrict__ b_psi,
                            const float* __restrict__ W_alp, const float* __restrict__ b_alp,
                            float* __restrict__ phiT, float* __restrict__ psiT,
                            float* __restrict__ alphaT) {
  extern __shared__ char smem[];
  _Float16* Wt = (_Float16*)smem;
  _Float16* Wf = Wt + 4096;
  _Float16* Ws = Wf + 4096;
  _Float16* Wa = Ws + 4096;
  float* par = (float*)(Wa + 4096);        // 6 x 64 floats
  _Float16* xb = (_Float16*)(par + 384);   // 8 waves x [16][64] f16
  int tid = threadIdx.x, lane = tid & 31, wv = tid >> 5;

  for (int i = tid; i < 4096; i += 256) {
    Wt[i] = (_Float16)W_top[i];
    Wf[i] = (_Float16)W_phi[i];
    Ws[i] = (_Float16)W_psi[i];
    Wa[i] = (_Float16)W_alp[i];
  }
  if (tid < 64) {
    par[tid]       = b_top[tid];
    par[64 + tid]  = s_top[tid];
    par[128 + tid] = t_top[tid];
    par[192 + tid] = b_phi[tid];
    par[256 + tid] = b_psi[tid];
    par[320 + tid] = b_alp[tid];
  }
  __syncthreads();

  int tile = blockIdx.x * 8 + wv;          // 0..1023
  int b = tile >> 9;
  int n0 = (tile & 511) << 4;
  _Float16* xw = xb + wv * 1024;           // [16][64] column-major
  int nl = lane & 15, ch = lane >> 4;
  int off = (lane & 16) ? 8 : 0;

  // stage raw x tile (f32 -> f16, column-major)
  for (int c0 = 0; c0 < 64; c0 += 2) {
    int c = c0 + ch;
    xw[nl * 64 + c] = (_Float16)x_in[((size_t)b * CH + c) * NPTS + n0 + nl];
  }
  LDS_FENCE();

  // x = s_top*(W_top@x + b_top) + t_top
  v8f acc[4] = {vzero8(), vzero8(), vzero8(), vzero8()};
  gemm64x16(Wt, xw, 0, lane, acc);
  LDS_FENCE();
#pragma unroll
  for (int rt = 0; rt < 4; ++rt) {
    int rb = rt * 16 + off;
    _Float16* pp = xw + nl * 64 + rb;
#pragma unroll
    for (int v = 0; v < 8; ++v) {
      int c = rb + v;
      pp[v] = (_Float16)(par[64 + c] * (acc[rt][v] + par[c]) + par[128 + c]);
    }
  }
  LDS_FENCE();

  size_t rowbase = ((size_t)b * NPTS + n0 + nl) * 64;
  // phi
  {
    v8f a2[4] = {vzero8(), vzero8(), vzero8(), vzero8()};
    gemm64x16(Wf, xw, 0, lane, a2);
#pragma unroll
    for (int rt = 0; rt < 4; ++rt) {
      int rb = rt * 16 + off;
#pragma unroll
      for (int v = 0; v < 8; ++v) phiT[rowbase + rb + v] = a2[rt][v] + par[192 + rb + v];
    }
  }
  // psi
  {
    v8f a2[4] = {vzero8(), vzero8(), vzero8(), vzero8()};
    gemm64x16(Ws, xw, 0, lane, a2);
#pragma unroll
    for (int rt = 0; rt < 4; ++rt) {
      int rb = rt * 16 + off;
#pragma unroll
      for (int v = 0; v < 8; ++v) psiT[rowbase + rb + v] = a2[rt][v] + par[256 + rb + v];
    }
  }
  // alpha
  {
    v8f a2[4] = {vzero8(), vzero8(), vzero8(), vzero8()};
    gemm64x16(Wa, xw, 0, lane, a2);
#pragma unroll
    for (int rt = 0; rt < 4; ++rt) {
      int rb = rt * 16 + off;
#pragma unroll
      for (int v = 0; v < 8; ++v) alphaT[rowbase + rb + v] = a2[rt][v] + par[320 + rb + v];
    }
  }
}

// ---------------------------------------------------------------------------
// Kernel 3: fused pos/pre/g/softmax/y/out + attn write.
// Block = 16 points (256 (n,k) columns, 8 waves, 2 column-tiles per wave).
// One 16-col WMMA tile == one point's 16 neighbors -> softmax is a width-16
// shuffle butterfly.
// ---------------------------------------------------------------------------
__global__ void fused_kernel(const float* __restrict__ p, const float* __restrict__ x_in,
                             const int* __restrict__ idxw,
                             const float* __restrict__ phiT, const float* __restrict__ psiT,
                             const float* __restrict__ alphaT,
                             const float* __restrict__ W_d1, const float* __restrict__ b_d1,
                             const float* __restrict__ s_d1, const float* __restrict__ t_d1,
                             const float* __restrict__ W_d2, const float* __restrict__ b_d2,
                             const float* __restrict__ s_d2, const float* __restrict__ t_d2,
                             const float* __restrict__ W_g1, const float* __restrict__ b_g1,
                             const float* __restrict__ s_g1, const float* __restrict__ t_g1,
                             const float* __restrict__ W_g2, const float* __restrict__ b_g2,
                             const float* __restrict__ s_g2, const float* __restrict__ t_g2,
                             const float* __restrict__ W_dn, const float* __restrict__ b_dn,
                             const float* __restrict__ s_dn, const float* __restrict__ t_dn,
                             float* __restrict__ outp, float* __restrict__ attnp) {
  extern __shared__ char smem[];
  _Float16* Wd2h = (_Float16*)smem;
  _Float16* Wg1h = Wd2h + 4096;
  _Float16* Wg2h = Wg1h + 4096;
  _Float16* Wdh  = Wg2h + 4096;
  float* Wd1f = (float*)(Wdh + 4096);      // [64*3]
  float* par  = Wd1f + 192;                // 15 x 64
  float* phis = par + 960;                 // [16][64]
  float* pn   = phis + 1024;               // [16][3]
  int*   midx = (int*)(pn + 48);           // [256]
  _Float16* colA = (_Float16*)(midx + 256);  // [256][64] h1, later alpha
  _Float16* posb = colA + 16384;             // [256][64]
  _Float16* preb = posb + 16384;             // [256][64]
  _Float16* ybuf = preb + 16384;             // [16][64]

  float* Pbd1 = par;        float* Psd1 = par + 64;  float* Ptd1 = par + 128;
  float* Pbd2 = par + 192;  float* Psd2 = par + 256; float* Ptd2 = par + 320;
  float* Pbg1 = par + 384;  float* Psg1 = par + 448; float* Ptg1 = par + 512;
  float* Pbg2 = par + 576;  float* Psg2 = par + 640; float* Ptg2 = par + 704;
  float* Pbdn = par + 768;  float* Psdn = par + 832; float* Ptdn = par + 896;

  int tid = threadIdx.x, lane = tid & 31, wv = tid >> 5;
  int b  = blockIdx.x >> 9;                // / (N/16)
  int n0 = (blockIdx.x & 511) << 4;

  for (int i = tid; i < 4096; i += 256) {
    Wd2h[i] = (_Float16)W_d2[i];
    Wg1h[i] = (_Float16)W_g1[i];
    Wg2h[i] = (_Float16)W_g2[i];
    Wdh[i]  = (_Float16)W_dn[i];
  }
  if (tid < 192) Wd1f[tid] = W_d1[tid];
  if (tid < 64) {
    Pbd1[tid] = b_d1[tid]; Psd1[tid] = s_d1[tid]; Ptd1[tid] = t_d1[tid];
    Pbd2[tid] = b_d2[tid]; Psd2[tid] = s_d2[tid]; Ptd2[tid] = t_d2[tid];
    Pbg1[tid] = b_g1[tid]; Psg1[tid] = s_g1[tid]; Ptg1[tid] = t_g1[tid];
    Pbg2[tid] = b_g2[tid]; Psg2[tid] = s_g2[tid]; Ptg2[tid] = t_g2[tid];
    Pbdn[tid] = b_dn[tid]; Psdn[tid] = s_dn[tid]; Ptdn[tid] = t_dn[tid];
  }
  for (int i = tid; i < 1024; i += 256)
    phis[i] = phiT[((size_t)b * NPTS + n0 + (i >> 6)) * 64 + (i & 63)];
  if (tid < 48) pn[tid] = p[(size_t)b * 3 * NPTS + (tid % 3) * NPTS + n0 + tid / 3];
  midx[tid] = idxw[((size_t)b * NPTS + n0 + (tid >> 4)) * KNN + (tid & 15)];
  __syncthreads();

  // P1: h1 = relu(bn_d1(W_d1 @ rel)), one (n,k) column per thread
  {
    int m = midx[tid];
    int pt = tid >> 4;
    const float* pb = p + (size_t)b * 3 * NPTS;
    float r0 = pn[pt * 3 + 0] - pb[m];
    float r1 = pn[pt * 3 + 1] - pb[NPTS + m];
    float r2 = pn[pt * 3 + 2] - pb[2 * NPTS + m];
    _Float16* h1p = colA + tid * 64;
    for (int c = 0; c < 64; ++c) {
      float h = Wd1f[c * 3] * r0 + Wd1f[c * 3 + 1] * r1 + Wd1f[c * 3 + 2] * r2 + Pbd1[c];
      h = Psd1[c] * h + Ptd1[c];
      h1p[c] = (_Float16)fmaxf(h, 0.f);
    }
  }
  LDS_FENCE();

  int nl = lane & 15;
  int off = (lane & 16) ? 8 : 0;

  for (int ti = 0; ti < 2; ++ti) {
    int ct  = wv * 2 + ti;           // column-tile == point
    int pt  = ct;
    int col = ct * 16 + nl;
    int n   = n0 + pt;

    // pos = bn_d2(W_d2 @ h1)
    v8f acc[4] = {vzero8(), vzero8(), vzero8(), vzero8()};
    gemm64x16(Wd2h, colA, ct * 16, lane, acc);
    LDS_FENCE();
#pragma unroll
    for (int rt = 0; rt < 4; ++rt) {
      int rb = rt * 16 + off;
      _Float16* pp = posb + col * 64 + rb;
#pragma unroll
      for (int v = 0; v < 8; ++v) {
        int c = rb + v;
        pp[v] = (_Float16)(Psd2[c] * (acc[rt][v] + Pbd2[c]) + Ptd2[c]);
      }
    }
    LDS_FENCE();

    // pre = phi - psi_g + pos ; stage alpha_g (reuse colA, own columns only)
    {
      int m = midx[col];
      const float* psc = psiT   + ((size_t)b * NPTS + m) * 64;
      const float* alc = alphaT + ((size_t)b * NPTS + m) * 64;
#pragma unroll
      for (int kc = 0; kc < 2; ++kc) {
        int cb = kc * 32 + ((lane & 16) ? 16 : 0);
#pragma unroll
        for (int j = 0; j < 16; ++j) {
          int c = cb + j;
          float pr = phis[pt * 64 + c] - psc[c] + (float)posb[col * 64 + c];
          preb[col * 64 + c] = (_Float16)pr;
          colA[col * 64 + c] = (_Float16)alc[c];
        }
      }
    }
    LDS_FENCE();

    // h2 = relu(bn_g1(W_g1 @ pre)) (overwrite preb in place, own columns)
    v8f acc2[4] = {vzero8(), vzero8(), vzero8(), vzero8()};
    gemm64x16(Wg1h, preb, ct * 16, lane, acc2);
    LDS_FENCE();
#pragma unroll
    for (int rt = 0; rt < 4; ++rt) {
      int rb = rt * 16 + off;
      _Float16* pp = preb + col * 64 + rb;
#pragma unroll
      for (int v = 0; v < 8; ++v) {
        int c = rb + v;
        float h = Psg1[c] * (acc2[rt][v] + Pbg1[c]) + Ptg1[c];
        pp[v] = (_Float16)fmaxf(h, 0.f);
      }
    }
    LDS_FENCE();

    // g = bn_g2(W_g2 @ h2)
    v8f g[4] = {vzero8(), vzero8(), vzero8(), vzero8()};
    gemm64x16(Wg2h, preb, ct * 16, lane, g);

    // softmax over k (16 lanes of each half) + attn write + y reduction
    int k = nl;
#pragma unroll
    for (int rt = 0; rt < 4; ++rt) {
      int rb = rt * 16 + off;
#pragma unroll
      for (int v = 0; v < 8; ++v) {
        int c = rb + v;
        float gv = Psg2[c] * (g[rt][v] + Pbg2[c]) + Ptg2[c];
        float mx = gv;
        mx = fmaxf(mx, __shfl_xor(mx, 1, 16));
        mx = fmaxf(mx, __shfl_xor(mx, 2, 16));
        mx = fmaxf(mx, __shfl_xor(mx, 4, 16));
        mx = fmaxf(mx, __shfl_xor(mx, 8, 16));
        float e = __expf(gv - mx);
        float sm = e;
        sm += __shfl_xor(sm, 1, 16);
        sm += __shfl_xor(sm, 2, 16);
        sm += __shfl_xor(sm, 4, 16);
        sm += __shfl_xor(sm, 8, 16);
        float a = e / sm;
        attnp[(((size_t)b * CH + c) * NPTS + n) * KNN + k] = a;
        float yv = a * ((float)colA[col * 64 + c] + (float)posb[col * 64 + c]);
        yv += __shfl_xor(yv, 1, 16);
        yv += __shfl_xor(yv, 2, 16);
        yv += __shfl_xor(yv, 4, 16);
        yv += __shfl_xor(yv, 8, 16);
        if (nl == 0) ybuf[pt * 64 + c] = (_Float16)yv;
      }
    }
  }
  __syncthreads();

  // out = bn_down(W_down @ y) + x_in  (one 64x16 GEMM for the whole block)
  if (wv == 0) {
    v8f acc[4] = {vzero8(), vzero8(), vzero8(), vzero8()};
    gemm64x16(Wdh, ybuf, 0, lane, acc);
    int n = n0 + nl;
#pragma unroll
    for (int rt = 0; rt < 4; ++rt) {
      int rb = rt * 16 + off;
#pragma unroll
      for (int v = 0; v < 8; ++v) {
        int c = rb + v;
        float o = Psdn[c] * (acc[rt][v] + Pbdn[c]) + Ptdn[c]
                + x_in[((size_t)b * CH + c) * NPTS + n];
        outp[((size_t)b * CH + c) * NPTS + n] = o;
      }
    }
  }
}

// ---------------------------------------------------------------------------
extern "C" void kernel_launch(void* const* d_in, const int* in_sizes, int n_in,
                              void* d_out, int out_size, void* d_ws, size_t ws_size,
                              hipStream_t stream) {
  const float* input_p = (const float*)d_in[0];
  const float* input_x = (const float*)d_in[1];
  const float* W_top = (const float*)d_in[2];
  const float* b_top = (const float*)d_in[3];
  const float* s_top = (const float*)d_in[4];
  const float* t_top = (const float*)d_in[5];
  const float* W_phi = (const float*)d_in[6];
  const float* b_phi = (const float*)d_in[7];
  const float* W_psi = (const float*)d_in[8];
  const float* b_psi = (const float*)d_in[9];
  const float* W_alp = (const float*)d_in[10];
  const float* b_alp = (const float*)d_in[11];
  const float* W_g1 = (const float*)d_in[12];
  const float* b_g1 = (const float*)d_in[13];
  const float* s_g1 = (const float*)d_in[14];
  const float* t_g1 = (const float*)d_in[15];
  const float* W_g2 = (const float*)d_in[16];
  const float* b_g2 = (const float*)d_in[17];
  const float* s_g2 = (const float*)d_in[18];
  const float* t_g2 = (const float*)d_in[19];
  const float* W_d1 = (const float*)d_in[20];
  const float* b_d1 = (const float*)d_in[21];
  const float* s_d1 = (const float*)d_in[22];
  const float* t_d1 = (const float*)d_in[23];
  const float* W_d2 = (const float*)d_in[24];
  const float* b_d2 = (const float*)d_in[25];
  const float* s_d2 = (const float*)d_in[26];
  const float* t_d2 = (const float*)d_in[27];
  const float* W_dn = (const float*)d_in[28];
  const float* b_dn = (const float*)d_in[29];
  const float* s_dn = (const float*)d_in[30];
  const float* t_dn = (const float*)d_in[31];

  float* outp  = (float*)d_out;
  float* attnp = outp + (size_t)BATCH * CH * NPTS;

  // workspace: idx [B,N,16] i32 | phiT | psiT | alphaT [B,N,64] f32 each (~13.6 MB)
  int*   idxw   = (int*)d_ws;
  float* phiT   = (float*)((char*)d_ws + (size_t)BATCH * NPTS * KNN * sizeof(int));
  float* psiT   = phiT + (size_t)BATCH * NPTS * CH;
  float* alphaT = psiT + (size_t)BATCH * NPTS * CH;

  size_t knnSmem  = (size_t)3 * NPTS * sizeof(float);                       // 98304
  size_t featSmem = 4 * 4096 * sizeof(_Float16) + 6 * 64 * sizeof(float)
                  + 8 * 16 * 64 * sizeof(_Float16);                         // 50688
  size_t fusedSmem = 4 * 4096 * sizeof(_Float16)
                   + (192 + 960 + 1024 + 48) * sizeof(float)
                   + 256 * sizeof(int)
                   + (3 * 256 * 64 + 16 * 64) * sizeof(_Float16);           // 143040

  knn_kernel<<<BATCH * 32, 256, knnSmem, stream>>>(input_p, idxw);
  feat_kernel<<<BATCH * (NPTS / 16) / 8, 256, featSmem, stream>>>(
      input_x, W_top, b_top, s_top, t_top, W_phi, b_phi, W_psi, b_psi,
      W_alp, b_alp, phiT, psiT, alphaT);
  fused_kernel<<<BATCH * (NPTS / 16), 256, fusedSmem, stream>>>(
      input_p, input_x, idxw, phiT, psiT, alphaT,
      W_d1, b_d1, s_d1, t_d1, W_d2, b_d2, s_d2, t_d2,
      W_g1, b_g1, s_g1, t_g1, W_g2, b_g2, s_g2, t_g2,
      W_dn, b_dn, s_dn, t_dn, outp, attnp);
}
